// SimpleTransformer_59871844106416
// MI455X (gfx1250) — compile-verified, matrix operand order checked by reading, not run on hardware
//
#include <hip/hip_runtime.h>
#include <hip/hip_bf16.h>
#include <math.h>
#include <stdint.h>

// ---------------------------------------------------------------- types
typedef __attribute__((ext_vector_type(16))) _Float16 v16h;
typedef __attribute__((ext_vector_type(8)))  _Float16 v8h;
typedef __attribute__((ext_vector_type(8)))  float    v8f;
typedef __attribute__((ext_vector_type(4)))  unsigned int u32x4;
typedef __attribute__((ext_vector_type(8)))  int          i32x8;
typedef __attribute__((ext_vector_type(4)))  int          i32x4;

typedef int v4i __attribute__((vector_size(16)));
typedef v4i __attribute__((address_space(1))) as1_v4i;
typedef v4i __attribute__((address_space(3))) as3_v4i;

#define NNODES 4096          // B * 256
#define NBR    16
#define HDIM   128
#define SROWS  (NBR * NNODES)
#define LDS_PAD 8            // halfs of padding per LDS row (= 4 DWORDs)

// ---------------------------------------------------------------- CDNA5 feature detection (graceful fallback)
#ifndef __has_builtin
#define __has_builtin(x) 0
#endif
#ifndef __has_include
#define __has_include(x) 0
#endif

#if __has_builtin(__builtin_amdgcn_global_load_async_to_lds_b128) && \
    __has_builtin(__builtin_amdgcn_s_wait_asynccnt)
#define HAVE_ASYNC_LDS 1
#else
#define HAVE_ASYNC_LDS 0
#endif

#if __has_builtin(__builtin_amdgcn_tensor_load_to_lds) && \
    __has_builtin(__builtin_amdgcn_s_wait_tensorcnt)
#define HAVE_TDM 1
#if __has_include(<hip/amd_detail/amd_gfx1250_TDM.h>)
#define TDM_6ARG 1            // therock-10.0 headers -> 6-arg builtin
#else
#define TDM_6ARG 0            // ROCm 7.2 -> 5-arg builtin
#endif
#else
#define HAVE_TDM 0
#endif

// ---------------------------------------------------------------- helpers
__device__ __forceinline__ float gelu_exact(float x) {
    return 0.5f * x * (1.0f + erff(x * 0.70710678118654752f));
}

__device__ __forceinline__ as1_v4i* to_glb_v4(const void* p) {
    return (as1_v4i*)(unsigned long long)(uintptr_t)p;
}
__device__ __forceinline__ as3_v4i* to_lds_v4(void* p) {
    // low 32 bits of the generic (LDS-aperture) address == LDS byte offset
    return (as3_v4i*)(unsigned int)(uintptr_t)p;
}

// TDM: DMA a 2D tile (64 rows x K halfs, row stride K halfs in global) into LDS,
// with hardware padding of 4 DWORDs (LDS_PAD halfs) after each row so the LDS
// layout matches our bank-conflict-free (K + LDS_PAD) stride.
__device__ __forceinline__ void tdm_load_tile(const _Float16* gsrc, _Float16* ldsDst, int K)
{
#if HAVE_TDM
    unsigned long long ga = (unsigned long long)(uintptr_t)gsrc;
    unsigned int ldsOff   = (unsigned int)(uintptr_t)ldsDst;
    // pad_interval code: row = K/2 DWORDs -> 2<<code == K/2
    int rowDW = K >> 1;
    int icode = (rowDW == 16) ? 3 : (rowDW == 32) ? 4 : 5;   // K = 32 / 64 / 128
    u32x4 g0;
    g0[0] = 1u;                                              // count = 1 (valid)
    g0[1] = ldsOff;                                          // lds_addr
    g0[2] = (unsigned int)ga;                                // global_addr[31:0]
    g0[3] = (unsigned int)((ga >> 32) & 0x01FFFFFFu) | 0x80000000u; // addr[56:32] | type=2
    int w0 = (1 << 16)            // data_size = 1 (2 bytes)
           | (1 << 20)            // pad_enable
           | (icode << 22)        // pad_interval
           | (3 << 25);           // pad_amount = 4 DWORDs
    i32x8 g1;
    g1[0] = w0;
    g1[1] = (int)((unsigned)K << 16);    // tensor_dim0[15:0]
    g1[2] = (int)(64u << 16);            // tensor_dim0 hi = 0 | tensor_dim1 lo = 64
    g1[3] = (int)((unsigned)K << 16);    // tensor_dim1 hi = 0 | tile_dim0 = K
    g1[4] = 64;                          // tile_dim1 = 64, tile_dim2 = 0
    g1[5] = K;                           // tensor_dim0_stride[31:0] = K
    g1[6] = 0;
    g1[7] = 0;
    i32x4 gz; gz[0] = gz[1] = gz[2] = gz[3] = 0;
#if TDM_6ARG
    i32x8 gz8; for (int i = 0; i < 8; ++i) gz8[i] = 0;
    __builtin_amdgcn_tensor_load_to_lds(g0, g1, gz, gz, gz8, 0);
#else
    __builtin_amdgcn_tensor_load_to_lds(g0, g1, gz, gz, 0);
#endif
#else
    (void)gsrc; (void)ldsDst; (void)K;
#endif
}

// ---------------------------------------------------------------- cvt / pad f32 -> f16
__global__ void __launch_bounds__(256)
cvt_pad_f16(const float* __restrict__ src, _Float16* __restrict__ dst,
            int rows, int cols, int colsPad)
{
    long i = (long)blockIdx.x * 256 + threadIdx.x;
    long total = (long)rows * colsPad;
    if (i >= total) return;
    int r = (int)(i / colsPad);
    int c = (int)(i % colsPad);
    float v = (c < cols) ? src[(size_t)r * cols + c] : 0.0f;
    dst[i] = (_Float16)v;
}

// ---------------------------------------------------------------- neighbor gather (+edge concat) -> f16
__global__ void __launch_bounds__(128)
gather_kernel(const int* __restrict__ adj, const float* __restrict__ src, int sd,
              const float* __restrict__ edges, int ev,
              _Float16* __restrict__ Ag, int Kp, int S)
{
    int row = blockIdx.x * 8 + (threadIdx.x >> 4);
    int lc  = threadIdx.x & 15;
    if (row >= S) return;
    int idx = adj[row];
    const float* srow = (idx > 0) ? (src + (size_t)(idx - 1) * sd) : nullptr;
    for (int c = lc; c < Kp; c += 16) {
        float v = 0.0f;
        if (c < sd) {
            if (srow) v = srow[c];
        } else if (edges && c < sd + ev) {
            v = edges[(size_t)row * ev + (c - sd)];
        }
        Ag[(size_t)row * Kp + c] = (_Float16)v;
    }
}

// ---------------------------------------------------------------- WMMA GEMM: Out = act(A[M x K] * W[Nc x K]^T + bias)
// grid.x = M/64, grid.y = Nc/64 ; 128 threads = 4 waves, each wave: 16 rows x 64 cols
// A tile streamed via GLOBAL_LOAD_ASYNC_TO_LDS_B128, W tile via TDM (with fallbacks).
template<bool GELU_ACT, bool OUT_F16>
__global__ void __launch_bounds__(128)
gemm_wmma(const _Float16* __restrict__ A, const _Float16* __restrict__ W,
          const float* __restrict__ bias, void* __restrict__ Out,
          int K, int ldo)
{
    __shared__ _Float16 As[64 * (128 + LDS_PAD)];
    __shared__ _Float16 Ws[64 * (128 + LDS_PAD)];
    const int ldL   = K + LDS_PAD;
    const int tid   = threadIdx.x;
    const int m_blk = blockIdx.x * 64;
    const int n_blk = blockIdx.y * 64;

#if HAVE_TDM
    if (tid == 0)
        tdm_load_tile(&W[(size_t)n_blk * K], &Ws[0], K);   // one TDM op per block
#endif

    // stage A (and W when no TDM); 16B chunks, uniform trip count (K multiple of 32)
    const int chunksPerRow = K >> 3;
    const int totalChunks  = 64 * chunksPerRow;
    for (int c = tid; c < totalChunks; c += 128) {
        int r  = c / chunksPerRow;
        int cc = (c - r * chunksPerRow) << 3;
        const _Float16* gA = &A[(size_t)(m_blk + r) * K + cc];
        _Float16*       lA = &As[r * ldL + cc];
#if HAVE_ASYNC_LDS
        __builtin_amdgcn_global_load_async_to_lds_b128(to_glb_v4(gA), to_lds_v4(lA), 0, 0);
#else
        *(v8h*)lA = *(const v8h*)gA;
#endif
#if !HAVE_TDM
        const _Float16* gW = &W[(size_t)(n_blk + r) * K + cc];
        _Float16*       lW = &Ws[r * ldL + cc];
#if HAVE_ASYNC_LDS
        __builtin_amdgcn_global_load_async_to_lds_b128(to_glb_v4(gW), to_lds_v4(lW), 0, 0);
#else
        *(v8h*)lW = *(const v8h*)gW;
#endif
#endif
    }
#if HAVE_ASYNC_LDS
    __builtin_amdgcn_s_wait_asynccnt(0);
#endif
#if HAVE_TDM
    __builtin_amdgcn_s_wait_tensorcnt(0);   // no-op for waves with TENSORcnt == 0
#endif
    __syncthreads();

    const int wave = tid >> 5;
    const int lane = tid & 31;
    const int half = lane >> 4;
    const int l16  = lane & 15;

    v8f acc[4];
#pragma unroll
    for (int nt = 0; nt < 4; ++nt)
#pragma unroll
        for (int r = 0; r < 8; ++r) acc[nt][r] = 0.0f;

    for (int kk = 0; kk < K; kk += 32) {
        // A fragment: lane holds row (wave*16+l16); v0..3 <- K[kk+8*half ..+8), v4..7 <- +16
        v16h a;
        {
            const _Float16* base = &As[(wave * 16 + l16) * ldL + kk + 8 * half];
            v8h lo = *(const v8h*)base;
            v8h hi = *(const v8h*)(base + 16);
#pragma unroll
            for (int i = 0; i < 8; ++i) { a[i] = lo[i]; a[8 + i] = hi[i]; }
        }
#pragma unroll
        for (int nt = 0; nt < 4; ++nt) {
            v16h b;
            const _Float16* base = &Ws[(nt * 16 + l16) * ldL + kk + 8 * half];
            v8h lo = *(const v8h*)base;
            v8h hi = *(const v8h*)(base + 16);
#pragma unroll
            for (int i = 0; i < 8; ++i) { b[i] = lo[i]; b[8 + i] = hi[i]; }
            acc[nt] = __builtin_amdgcn_wmma_f32_16x16x32_f16(
                false, a, false, b, (short)0, acc[nt], false, false);
        }
    }

    // epilogue: D layout — n = lane%16, vgpr r => m = r + 8*(lane/16)
#pragma unroll
    for (int nt = 0; nt < 4; ++nt) {
        int col  = n_blk + nt * 16 + l16;
        float bv = bias ? bias[col] : 0.0f;
#pragma unroll
        for (int r = 0; r < 8; ++r) {
            int row = m_blk + wave * 16 + half * 8 + r;
            float v = acc[nt][r] + bv;
            if (GELU_ACT) v = gelu_exact(v);
            if (OUT_F16)
                ((_Float16*)Out)[(size_t)row * ldo + col] = (_Float16)v;
            else
                ((float*)Out)[(size_t)row * ldo + col] = v;
        }
    }
}

// ---------------------------------------------------------------- attention (q_len = 1): one node per block, one wave per head
__global__ void __launch_bounds__(128)
attn_kernel(const float* __restrict__ qh, const _Float16* __restrict__ kh,
            const _Float16* __restrict__ vh, const unsigned char* __restrict__ mask,
            _Float16* __restrict__ o)
{
    const int n = blockIdx.x;
    const int t = threadIdx.x;              // head = t/32 (== wave), dim lane = t%32
    float qv = qh[(size_t)n * HDIM + t];

    float sc[NBR];
#pragma unroll
    for (int j = 0; j < NBR; ++j) {
        float kval = (float)kh[((size_t)j * NNODES + n) * HDIM + t];
        float prod = qv * kval;
#pragma unroll
        for (int off = 16; off >= 1; off >>= 1) prod += __shfl_xor(prod, off, 32);
        sc[j] = prod * 0.17677669529663688f;   // 1/sqrt(32)
    }
    float mx = -1e30f;
#pragma unroll
    for (int j = 0; j < NBR; ++j) {
        if (mask[(size_t)n * NBR + j]) sc[j] = -1e30f;
        mx = fmaxf(mx, sc[j]);
    }
    float den = 0.0f;
#pragma unroll
    for (int j = 0; j < NBR; ++j) { sc[j] = __expf(sc[j] - mx); den += sc[j]; }
    float inv = 1.0f / den;
    float accv = 0.0f;
#pragma unroll
    for (int j = 0; j < NBR; ++j)
        accv += sc[j] * inv * (float)vh[((size_t)j * NNODES + n) * HDIM + t];
    o[(size_t)n * HDIM + t] = (_Float16)accv;
}

// ---------------------------------------------------------------- layernorm of (x + x): one row per block
__global__ void __launch_bounds__(128)
ln_kernel(const float* __restrict__ x, const float* __restrict__ g,
          const float* __restrict__ b,
          float* __restrict__ out1, int ldo1,
          float* __restrict__ out2, int ldo2)
{
    __shared__ float red[4];
    const int r = blockIdx.x, t = threadIdx.x;
    float v = 2.0f * x[(size_t)r * HDIM + t];     // o = o + o (residual in eval)

    float s = v;
#pragma unroll
    for (int off = 16; off >= 1; off >>= 1) s += __shfl_xor(s, off, 32);
    if ((t & 31) == 0) red[t >> 5] = s;
    __syncthreads();
    float mu = (red[0] + red[1] + red[2] + red[3]) * (1.0f / HDIM);
    float d  = v - mu;
    __syncthreads();

    float s2 = d * d;
#pragma unroll
    for (int off = 16; off >= 1; off >>= 1) s2 += __shfl_xor(s2, off, 32);
    if ((t & 31) == 0) red[t >> 5] = s2;
    __syncthreads();
    float var = (red[0] + red[1] + red[2] + red[3]) * (1.0f / HDIM);

    float y = d * rsqrtf(var + 1e-5f) * g[t] + b[t];
    out1[(size_t)r * ldo1 + t] = y;
    if (out2) out2[(size_t)r * ldo2 + t] = y;
}

// ---------------------------------------------------------------- host-side GAT driver
struct GatW {
    const float *b, *dest_W, *dest_b, *g, *in_W, *in_b, *out_W, *out_b, *src_W, *src_b;
};

struct Scratch {
    _Float16 *Ag, *kv, *kh, *vh, *q16, *o16;
    float *qh, *oproj;
};

static void run_gat(hipStream_t stream,
                    const float* src_f32, int sd, const float* edges, int ev, int Kp,
                    const _Float16* dest16, int dd,
                    const int* adj, const unsigned char* mask,
                    const _Float16* srcW16, const _Float16* destW16,
                    const _Float16* inW16,  const _Float16* outW16,
                    const GatW& w,
                    float* out1, int ldo1, float* out2, int ldo2,
                    const Scratch& s)
{
    gather_kernel<<<SROWS / 8, 128, 0, stream>>>(adj, src_f32, sd, edges, ev, s.Ag, Kp, SROWS);

    // kv = GELU(Ag @ srcW^T + src_b)          (65536 x 128, f16)
    gemm_wmma<true, true><<<dim3(SROWS / 64, 2), 128, 0, stream>>>(
        s.Ag, srcW16, w.src_b, s.kv, Kp, HDIM);
    // q = GELU(dest @ destW^T + dest_b)       (4096 x 128, f16)
    gemm_wmma<true, true><<<dim3(NNODES / 64, 2), 128, 0, stream>>>(
        dest16, destW16, w.dest_b, s.q16, dd, HDIM);
    // qh = q @ Wq^T + bq                      (4096 x 128, f32)
    gemm_wmma<false, false><<<dim3(NNODES / 64, 2), 128, 0, stream>>>(
        s.q16, inW16, w.in_b, s.qh, HDIM, HDIM);
    // kh = kv @ Wk^T + bk                     (65536 x 128, f16)
    gemm_wmma<false, true><<<dim3(SROWS / 64, 2), 128, 0, stream>>>(
        s.kv, inW16 + 128 * 128, w.in_b + 128, s.kh, HDIM, HDIM);
    // vh = kv @ Wv^T + bv
    gemm_wmma<false, true><<<dim3(SROWS / 64, 2), 128, 0, stream>>>(
        s.kv, inW16 + 256 * 128, w.in_b + 256, s.vh, HDIM, HDIM);

    attn_kernel<<<NNODES, 128, 0, stream>>>(s.qh, s.kh, s.vh, mask, s.o16);

    // oproj = o @ outW^T + out_b              (4096 x 128, f32)
    gemm_wmma<false, false><<<dim3(NNODES / 64, 2), 128, 0, stream>>>(
        s.o16, outW16, w.out_b, s.oproj, HDIM, HDIM);

    ln_kernel<<<NNODES, 128, 0, stream>>>(s.oproj, w.g, w.b, out1, ldo1, out2, ldo2);
}

static inline void launch_cvt(hipStream_t stream, const float* src, _Float16* dst,
                              int rows, int cols, int colsPad)
{
    long total = (long)rows * colsPad;
    cvt_pad_f16<<<(unsigned)((total + 255) / 256), 256, 0, stream>>>(src, dst, rows, cols, colsPad);
}

// ---------------------------------------------------------------- entry
extern "C" void kernel_launch(void* const* d_in, const int* in_sizes, int n_in,
                              void* d_out, int out_size, void* d_ws, size_t ws_size,
                              hipStream_t stream)
{
    (void)in_sizes; (void)n_in; (void)out_size; (void)ws_size;

    const float* p       = (const float*)d_in[0];
    const float* v       = (const float*)d_in[1];
    const int*   pe_adj  = (const int*)d_in[2];
    const unsigned char* pe_mask  = (const unsigned char*)d_in[3];
    const float* pe_edge = (const float*)d_in[4];
    const int*   ve_adj  = (const int*)d_in[5];
    const unsigned char* ve_mask  = (const unsigned char*)d_in[6];
    const float* ve_edge = (const float*)d_in[7];
    const int*   ppe_adj = (const int*)d_in[8];
    const unsigned char* ppe_mask = (const unsigned char*)d_in[9];
    const float* ppe_edge= (const float*)d_in[10];

    // params pytree order: gats sorted ('p0','p1','pp0','pp1','v0'),
    // leaves sorted ('b','dest_W','dest_b','g','in_W','in_b','out_W','out_b','src_W','src_b')
    GatW gw[5];
    for (int i = 0; i < 5; ++i) {
        int base = 11 + 10 * i;
        gw[i].b      = (const float*)d_in[base + 0];
        gw[i].dest_W = (const float*)d_in[base + 1];
        gw[i].dest_b = (const float*)d_in[base + 2];
        gw[i].g      = (const float*)d_in[base + 3];
        gw[i].in_W   = (const float*)d_in[base + 4];
        gw[i].in_b   = (const float*)d_in[base + 5];
        gw[i].out_W  = (const float*)d_in[base + 6];
        gw[i].out_b  = (const float*)d_in[base + 7];
        gw[i].src_W  = (const float*)d_in[base + 8];
        gw[i].src_b  = (const float*)d_in[base + 9];
    }
    const int IDX_P0 = 0, IDX_P1 = 1, IDX_PP0 = 2, IDX_PP1 = 3, IDX_V0 = 4;

    float* outF = (float*)d_out;
    float* outV1 = outF + (size_t)NNODES * 256;   // second output: v1 (4096 x 128)

    // ---------------- workspace carve-up
    char* ws = (char*)d_ws;
    size_t cur = 0;
    auto alloc = [&](size_t bytes) -> void* {
        void* ptr = ws + cur;
        cur = (cur + bytes + 255) & ~(size_t)255;
        return ptr;
    };

    float*    p1f  = (float*)alloc((size_t)NNODES * HDIM * 4);
    float*    v1f  = (float*)alloc((size_t)NNODES * HDIM * 4);
    float*    pp1f = (float*)alloc((size_t)NNODES * HDIM * 4);
    _Float16* p1h  = (_Float16*)alloc((size_t)NNODES * HDIM * 2);
    _Float16* pp1h = (_Float16*)alloc((size_t)NNODES * HDIM * 2);
    _Float16* p16  = (_Float16*)alloc((size_t)NNODES * 32 * 2);
    _Float16* v16  = (_Float16*)alloc((size_t)NNODES * 32 * 2);

    _Float16 *srcW16[5], *destW16[5], *inW16[5], *outW16[5];
    for (int i = 0; i < 5; ++i) {
        srcW16[i]  = (_Float16*)alloc((size_t)128 * 128 * 2);
        destW16[i] = (_Float16*)alloc((size_t)128 * 128 * 2);
        inW16[i]   = (_Float16*)alloc((size_t)384 * 128 * 2);
        outW16[i]  = (_Float16*)alloc((size_t)128 * 128 * 2);
    }

    Scratch s;
    s.Ag    = (_Float16*)alloc((size_t)SROWS * HDIM * 2);
    s.kv    = (_Float16*)alloc((size_t)SROWS * HDIM * 2);
    s.kh    = (_Float16*)alloc((size_t)SROWS * HDIM * 2);
    s.vh    = (_Float16*)alloc((size_t)SROWS * HDIM * 2);
    s.q16   = (_Float16*)alloc((size_t)NNODES * HDIM * 2);
    s.o16   = (_Float16*)alloc((size_t)NNODES * HDIM * 2);
    s.qh    = (float*)alloc((size_t)NNODES * HDIM * 4);
    s.oproj = (float*)alloc((size_t)NNODES * HDIM * 4);

    // ---------------- weight conversion (f32 -> f16, zero-pad K 48 -> 64 for layer 0 src GEMM)
    const int Kin[5] = {48, 128, 48, 128, 48};
    const int Kpd[5] = {64, 128, 64, 128, 64};
    const int dd [5] = {32, 128, 32, 128, 32};
    for (int i = 0; i < 5; ++i) {
        launch_cvt(stream, gw[i].src_W,  srcW16[i],  128, Kin[i], Kpd[i]);
        launch_cvt(stream, gw[i].dest_W, destW16[i], 128, dd[i],  dd[i]);
        launch_cvt(stream, gw[i].in_W,   inW16[i],   384, 128,    128);
        launch_cvt(stream, gw[i].out_W,  outW16[i],  128, 128,    128);
    }
    launch_cvt(stream, p, p16, NNODES, 32, 32);
    launch_cvt(stream, v, v16, NNODES, 32, 32);

    // ---------------- layer 0 (with edge features)
    // p1 = gat(src=v, dest=p)
    run_gat(stream, v, 32, pe_edge, 16, 64, p16, 32, pe_adj, pe_mask,
            srcW16[IDX_P0], destW16[IDX_P0], inW16[IDX_P0], outW16[IDX_P0], gw[IDX_P0],
            p1f, HDIM, nullptr, 0, s);
    // v1 = gat(src=p, dest=v) ; v1 is also the second model output
    run_gat(stream, p, 32, ve_edge, 16, 64, v16, 32, ve_adj, ve_mask,
            srcW16[IDX_V0], destW16[IDX_V0], inW16[IDX_V0], outW16[IDX_V0], gw[IDX_V0],
            v1f, HDIM, outV1, HDIM, s);
    // pp1 = gat(src=p, dest=p)
    run_gat(stream, p, 32, ppe_edge, 16, 64, p16, 32, ppe_adj, ppe_mask,
            srcW16[IDX_PP0], destW16[IDX_PP0], inW16[IDX_PP0], outW16[IDX_PP0], gw[IDX_PP0],
            pp1f, HDIM, nullptr, 0, s);

    launch_cvt(stream, p1f,  p1h,  NNODES, HDIM, HDIM);
    launch_cvt(stream, pp1f, pp1h, NNODES, HDIM, HDIM);

    // ---------------- layer 1 (no edges) — writes directly into d_out (B, PC, 256)
    // p2 = gat(src=v1, dest=p1)   -> cols [0,128)
    run_gat(stream, v1f, HDIM, nullptr, 0, HDIM, p1h, HDIM, pe_adj, pe_mask,
            srcW16[IDX_P1], destW16[IDX_P1], inW16[IDX_P1], outW16[IDX_P1], gw[IDX_P1],
            outF, 256, nullptr, 0, s);
    // pp2 = gat(src=pp1, dest=pp1) -> cols [128,256)
    run_gat(stream, pp1f, HDIM, nullptr, 0, HDIM, pp1h, HDIM, ppe_adj, ppe_mask,
            srcW16[IDX_PP1], destW16[IDX_PP1], inW16[IDX_PP1], outW16[IDX_PP1], gw[IDX_PP1],
            outF + 128, 256, nullptr, 0, s);
}
